// SphericalUNet_66546223284443
// MI455X (gfx1250) — matrix operand order repaired and checked by reading, compiled, not run on hardware
//
#include <hip/hip_runtime.h>
#include <hip/hip_bf16.h>

// ---------------------------------------------------------------------------
// Spherical U-Net forward on MI455X (gfx1250).
// Activations: [V, B, C] fp32 ("vertex-major") => icopool = prefix rows (free),
// GEMM rows m = v*B + b. GEMMs: V_WMMA_F32_16X16X32_BF16, A/W staged into LDS
// with GLOBAL_LOAD_ASYNC_TO_LDS_B128 (ASYNCcnt), double-buffered so stage i+1
// loads overlap stage i compute (s_wait_asynccnt 4). Fragments: ds_load_b128.
// K loop is segment-major (3 Chebyshev terms): scalar A base pointers,
// branch-free full stages, predicated zero-padded tail only when Cin%64 != 0.
// ---------------------------------------------------------------------------

typedef __attribute__((ext_vector_type(16))) __bf16 v16bf;
typedef __attribute__((ext_vector_type(8)))  float  v8f;

#define TM 64
#define TN 64
#define KSTEP 64
#define LDSSTRIDE 4096           // ushorts per (64 x 64) buffer
#define DEGP1 7
#define BN_EPS 1e-5f

static __device__ __forceinline__ unsigned short f32_bf16(float f) {
  union { float f; unsigned u; } cv; cv.f = f;
  unsigned u = cv.u;
  unsigned r = (u + 0x7fffu + ((u >> 16) & 1u)) >> 16;   // round-nearest-even
  return (unsigned short)r;
}

// gfx1250 async global->LDS copy (16B per lane), tracked by ASYNCcnt.
static __device__ __forceinline__ void async_g2l_b128(void* lds, const void* g) {
  unsigned l = (unsigned)(size_t)lds;
  asm volatile("global_load_async_to_lds_b128 %0, %1, off" :: "v"(l), "v"(g) : "memory");
}
static __device__ __forceinline__ void wait_async0() {
#if __has_builtin(__builtin_amdgcn_s_wait_asynccnt)
  __builtin_amdgcn_s_wait_asynccnt(0);
#else
  asm volatile("s_wait_asynccnt 0" ::: "memory");
#endif
}
static __device__ __forceinline__ void wait_async4() {   // cur buf done, next in flight
#if __has_builtin(__builtin_amdgcn_s_wait_asynccnt)
  __builtin_amdgcn_s_wait_asynccnt(4);
#else
  asm volatile("s_wait_asynccnt 4" ::: "memory");
#endif
}

// ---------------------------- elementwise kernels --------------------------

// weights fp32 [Kc, N] -> bf16 transposed [N, Kc]
__global__ __launch_bounds__(256) void k_wt_bf16(const float* __restrict__ w,
                                                 unsigned short* __restrict__ wt,
                                                 int Kc, int N) {
  int k = blockIdx.x * 256 + threadIdx.x;
  int n = blockIdx.y;
  if (k < Kc) wt[(size_t)n * Kc + k] = f32_bf16(w[(size_t)k * N + n]);
}

// x [B,V,C] -> y [V,B,C] fp32 + bf16
__global__ __launch_bounds__(256) void k_transpose_in(const float* __restrict__ x,
                                                      float* __restrict__ y,
                                                      unsigned short* __restrict__ yb,
                                                      int V, int B, int C) {
  int i = blockIdx.x * 256 + threadIdx.x;
  int n = V * B * C;
  if (i >= n) return;
  int c = i % C; int t = i / C; int b = t % B; int v = t / B;
  float r = x[((size_t)b * V + v) * C + c];
  y[i] = r;
  yb[i] = f32_bf16(r);
}

// float4-vectorized SpMM over [V, F] (F = B*C, multiple of 4):
// y[v,f] = alpha * sum_j vals[7v+j] * x[cols[7v+j], f] + beta * xprev[v,f]
__global__ __launch_bounds__(256) void k_spmm(const int* __restrict__ cols,
                                              const float* __restrict__ vals,
                                              const float* __restrict__ x,
                                              const float* __restrict__ xprev,
                                              float alpha, float beta,
                                              float* __restrict__ yf,
                                              unsigned short* __restrict__ ybf,
                                              int V, int F4) {
  int f4 = blockIdx.x * 256 + threadIdx.x;
  int v = blockIdx.y;
  if (f4 >= F4 || v >= V) return;
  const float4* x4 = (const float4*)x;
  int e = v * DEGP1;
  float ax = 0.f, ay = 0.f, az = 0.f, aw = 0.f;
#pragma unroll
  for (int j = 0; j < DEGP1; ++j) {
    float wv = vals[e + j];
    float4 xv = x4[(size_t)cols[e + j] * F4 + f4];
    ax += wv * xv.x; ay += wv * xv.y; az += wv * xv.z; aw += wv * xv.w;
  }
  float4 r;
  r.x = alpha * ax; r.y = alpha * ay; r.z = alpha * az; r.w = alpha * aw;
  size_t o = (size_t)v * F4 + f4;
  if (xprev) {
    float4 p = ((const float4*)xprev)[o];
    r.x += beta * p.x; r.y += beta * p.y; r.z += beta * p.z; r.w += beta * p.w;
  }
  if (yf) ((float4*)yf)[o] = r;
  union { unsigned short u[4]; uint2 q; } ob;
  ob.u[0] = f32_bf16(r.x); ob.u[1] = f32_bf16(r.y);
  ob.u[2] = f32_bf16(r.z); ob.u[3] = f32_bf16(r.w);
  *(uint2*)&ybf[o * 4] = ob.q;
}

// icounpool: rows >= Vsrc filled with 1.0; emits fp32 + bf16
__global__ __launch_bounds__(256) void k_unpool(const float* __restrict__ x,
                                                float* __restrict__ y,
                                                unsigned short* __restrict__ yb,
                                                int Vsrc, int Vdst, int F) {
  int i = blockIdx.x * 256 + threadIdx.x;
  int n = Vdst * F;
  if (i >= n) return;
  int v = i / F;
  float r = (v < Vsrc) ? x[i] : 1.0f;
  y[i] = r;
  yb[i] = f32_bf16(r);
}

// channel concat + bf16 emit
__global__ __launch_bounds__(256) void k_concat(const float* __restrict__ a,
                                                const float* __restrict__ bsrc,
                                                float* __restrict__ y,
                                                unsigned short* __restrict__ yb,
                                                int VB, int C1, int C2) {
  int C = C1 + C2;
  int i = blockIdx.x * 256 + threadIdx.x;
  int n = VB * C;
  if (i >= n) return;
  int c = i % C; int m = i / C;
  float r = (c < C1) ? a[(size_t)m * C1 + c] : bsrc[(size_t)m * C2 + (c - C1)];
  y[i] = r;
  yb[i] = f32_bf16(r);
}

// per-channel mean/biased-var over M rows of x [M,N]
__global__ __launch_bounds__(256) void k_bn_stats(const float* __restrict__ x,
                                                  float* __restrict__ stats,
                                                  int M, int N) {
  __shared__ float sh1[256], sh2[256];
  int n = blockIdx.x;
  float s = 0.f, s2 = 0.f;
  for (int m = threadIdx.x; m < M; m += 256) {
    float v = x[(size_t)m * N + n];
    s += v; s2 += v * v;
  }
  sh1[threadIdx.x] = s; sh2[threadIdx.x] = s2;
  __syncthreads();
  for (int off = 128; off > 0; off >>= 1) {
    if (threadIdx.x < off) {
      sh1[threadIdx.x] += sh1[threadIdx.x + off];
      sh2[threadIdx.x] += sh2[threadIdx.x + off];
    }
    __syncthreads();
  }
  if (threadIdx.x == 0) {
    float mean = sh1[0] / (float)M;
    stats[2 * n]     = mean;
    stats[2 * n + 1] = sh2[0] / (float)M - mean * mean;
  }
}

// BN(train) + ReLU, emits fp32 + bf16 (bf16 = x0 of the next layer)
__global__ __launch_bounds__(256) void k_bn_apply_relu(const float* __restrict__ x,
                                                       const float* __restrict__ stats,
                                                       const float* __restrict__ g,
                                                       const float* __restrict__ bt,
                                                       float* __restrict__ y,
                                                       unsigned short* __restrict__ yb,
                                                       int M, int N) {
  int i = blockIdx.x * 256 + threadIdx.x;
  int n = M * N;
  if (i >= n) return;
  int c = i % N;
  float mean = stats[2 * c], var = stats[2 * c + 1];
  float r = (x[i] - mean) * rsqrtf(var + BN_EPS) * g[c] + bt[c];
  r = r > 0.f ? r : 0.f;
  y[i] = r;
  yb[i] = f32_bf16(r);
}

// gin rows m = v*B + b, N cols -> out [B,V,N]
__global__ __launch_bounds__(256) void k_final_transpose(const float* __restrict__ gin,
                                                         float* __restrict__ out,
                                                         int V, int B, int N) {
  int i = blockIdx.x * 256 + threadIdx.x;
  int n = V * B * N;
  if (i >= n) return;
  int o = i % N; int t = i / N; int b = t % B; int v = t / B;
  out[((size_t)b * V + v) * N + o] = gin[i];
}

// ------------------------------- WMMA GEMM ---------------------------------
// out[M,N] = [A0|A1|A2][M,3*Cin](bf16) x Wt^T (Wt stored [N,Kc] bf16) + bias.
// 256 threads = 8 wave32s as 4(M) x 2(N); each wave holds two 16x16 accs
// (A-frag reused) => 4 WMMA per stage. Double-buffered async pipeline.
__global__ __launch_bounds__(256) void cheb_gemm_wmma(
    const unsigned short* __restrict__ A0, const unsigned short* __restrict__ A1,
    const unsigned short* __restrict__ A2, const unsigned short* __restrict__ Wt,
    const float* __restrict__ bias, float* __restrict__ out,
    int M, int N, int Cin) {
  __shared__ __attribute__((aligned(16))) unsigned short As[2][TM][KSTEP];  // [buf][m][k]
  __shared__ __attribute__((aligned(16))) unsigned short Bt[2][TN][KSTEP];  // [buf][n][k]
  const int Kc   = 3 * Cin;
  const int tid  = threadIdx.x;
  const int lane = tid & 31;
  const int wave = tid >> 5;
  const int wm = wave & 3, wn = wave >> 2;
  const int mb = blockIdx.y * TM, nb = blockIdx.x * TN;

  // A loader: 64 rows x (4 threads * 16 elems); clamp OOB rows (never stored)
  const int arow = tid >> 2, ac0 = (tid & 3) * 16;
  const int am = mb + arow;
  const size_t aoff = (size_t)(am < M ? am : (M - 1)) * Cin + ac0;
  unsigned short* adst = &As[0][arow][ac0];

  // B loader: 64 rows x (4 threads * 16 elems); Wt row-contiguous in K
  const int bnn = nb + arow;
  const unsigned short* wrow = Wt + (size_t)(bnn < N ? bnn : (N - 1)) * Kc + ac0;
  unsigned short* bdst = &Bt[0][arow][ac0];

  const int half = lane >> 4, lr = lane & 15;
  const unsigned short* afrow  = &As[0][wm * 16 + lr][0];
  const unsigned short* bfrow0 = &Bt[0][wn * 32 + lr][0];
  const unsigned short* bfrow1 = &Bt[0][wn * 32 + 16 + lr][0];

  v8f acc0 = {}, acc1 = {};

  const int Cfull = Cin & ~(KSTEP - 1);
  int buf = 0;

  // issue one stage's 4 async loads into LDS buffer b
  auto issue = [&](int b, const unsigned short* pa, const unsigned short* pw, int c0) {
    unsigned short* ad = adst + b * LDSSTRIDE;
    unsigned short* bd = bdst + b * LDSSTRIDE;
    async_g2l_b128(ad,     pa + c0);
    async_g2l_b128(ad + 8, pa + c0 + 8);
    async_g2l_b128(bd,     pw + c0);
    async_g2l_b128(bd + 8, pw + c0 + 8);
  };
  // 4 WMMAs over one 64-deep stage in LDS buffer b
  auto compute = [&](int b) {
    const unsigned short* ar = afrow  + b * LDSSTRIDE;
    const unsigned short* b0 = bfrow0 + b * LDSSTRIDE;
    const unsigned short* b1 = bfrow1 + b * LDSSTRIDE;
#pragma unroll
    for (int ks = 0; ks < 2; ++ks) {
      union { v16bf v; uint4 q[2]; } af, bf0, bf1;
      af.q[0]  = *(const uint4*)(ar + ks * 32 + half * 16);
      af.q[1]  = *(const uint4*)(ar + ks * 32 + half * 16 + 8);
      bf0.q[0] = *(const uint4*)(b0 + ks * 32 + half * 16);
      bf0.q[1] = *(const uint4*)(b0 + ks * 32 + half * 16 + 8);
      bf1.q[0] = *(const uint4*)(b1 + ks * 32 + half * 16);
      bf1.q[1] = *(const uint4*)(b1 + ks * 32 + half * 16 + 8);
      acc0 = __builtin_amdgcn_wmma_f32_16x16x32_bf16(false, af.v, false, bf0.v,
                                                     (short)0, acc0, false, false);
      acc1 = __builtin_amdgcn_wmma_f32_16x16x32_bf16(false, af.v, false, bf1.v,
                                                     (short)0, acc1, false, false);
    }
  };

  for (int seg = 0; seg < 3; ++seg) {
    const unsigned short* aseg = (seg == 0) ? A0 : ((seg == 1) ? A1 : A2);
    const unsigned short* pa = aseg + aoff;          // per-thread row, seg base
    const unsigned short* pw = wrow + seg * Cin;

    if (Cfull > 0) {
      issue(buf, pa, pw, 0);                         // prologue
      for (int c0 = 0; c0 < Cfull; c0 += KSTEP) {
        const int cn = c0 + KSTEP;
        const bool hn = cn < Cfull;
        if (hn) {
          issue(buf ^ 1, pa, pw, cn);                // overlap next stage's DMA
          __builtin_prefetch(pw + cn + KSTEP, 0, 0); // -> global_prefetch_b8
          wait_async4();                             // cur buf done, next in flight
        } else {
          wait_async0();
        }
        __syncthreads();
        compute(buf);
        __syncthreads();
        buf ^= 1;
      }
    }
    if (Cfull < Cin) {                               // tail: rem in {16,32,48}
      const int rem = Cin - Cfull;
      unsigned short* ad = adst + buf * LDSSTRIDE;
      unsigned short* bd = bdst + buf * LDSSTRIDE;
#pragma unroll
      for (int ch = 0; ch < 2; ++ch) {
        const int kk = ac0 + ch * 8;
        if (kk < rem) async_g2l_b128(ad + ch * 8, pa + Cfull + ch * 8);
        else          *(uint4*)(ad + ch * 8) = make_uint4(0u, 0u, 0u, 0u);
        if (kk < rem) async_g2l_b128(bd + ch * 8, pw + Cfull + ch * 8);
        else          *(uint4*)(bd + ch * 8) = make_uint4(0u, 0u, 0u, 0u);
      }
      wait_async0();
      __syncthreads();
      compute(buf);
      __syncthreads();
      buf ^= 1;
    }
  }

  const int n0 = nb + wn * 32 + lr;
  const int n1 = n0 + 16;
  const float bv0 = (bias && n0 < N) ? bias[n0] : 0.0f;
  const float bv1 = (bias && n1 < N) ? bias[n1] : 0.0f;
#pragma unroll
  for (int r = 0; r < 8; ++r) {
    int m = mb + wm * 16 + half * 8 + r;
    if (m < M) {
      if (n0 < N) out[(size_t)m * N + n0] = acc0[r] + bv0;
      if (n1 < N) out[(size_t)m * N + n1] = acc1[r] + bv1;
    }
  }
}

// ------------------------------- host driver -------------------------------

static inline int ceil_div(int a, int b) { return (a + b - 1) / b; }

namespace {
struct ChebTemps {
  float* t1;                 // fp32 x1 for the recurrence
  float* g;                  // fp32 pre-BN gemm output
  unsigned short *a0, *a1, *a2;
  float* stats;
};

// one Chebyshev conv (+optional BN-ReLU). actin fp32 [V,B,Cin]; a_in_bf = bf16 x0.
// If BN: writes actout fp32 and T.a0 bf16 (x0 of the next conv).
void run_cheb(hipStream_t s, const float* actin, const unsigned short* a_in_bf,
              int V, int B, int Cin, int Cout,
              const int* cols, const float* vals,
              const unsigned short* Wt, const float* bias,
              const float* gamma, const float* beta,
              float* actout, const ChebTemps& T) {
  const int F = B * Cin, M = V * B;
  dim3 sg(ceil_div(F / 4, 256), V);
  k_spmm<<<sg, 256, 0, s>>>(cols, vals, actin, nullptr, 1.f, 0.f, T.t1, T.a1, V, F / 4);
  k_spmm<<<sg, 256, 0, s>>>(cols, vals, T.t1, actin, 2.f, -1.f, nullptr, T.a2, V, F / 4);
  float* gout = gamma ? T.g : actout;
  dim3 gg(ceil_div(Cout, TN), ceil_div(M, TM));
  cheb_gemm_wmma<<<gg, 256, 0, s>>>(a_in_bf, T.a1, T.a2, Wt, bias, gout, M, Cout, Cin);
  if (gamma) {
    k_bn_stats<<<Cout, 256, 0, s>>>(gout, T.stats, M, Cout);
    k_bn_apply_relu<<<ceil_div(M * Cout, 256), 256, 0, s>>>(gout, T.stats, gamma, beta,
                                                            actout, T.a0, M, Cout);
  }
}
}  // namespace

extern "C" void kernel_launch(void* const* d_in, const int* in_sizes, int n_in,
                              void* d_out, int out_size, void* d_ws, size_t ws_size,
                              hipStream_t stream) {
  (void)in_sizes; (void)n_in; (void)out_size; (void)ws_size;
  const int B = 8;
  static const int VL[6] = {42, 162, 642, 2562, 10242, 40962};
  // LAYER_SPECS order: e5a,e5b,e4,e3,e2,e1,e0,d1a,d1b,d2a,d2b,d3a,d3b,d4a,d4b,d5a,d5b
  static const int spec[17][3] = {
      {16, 32, 1},  {32, 64, 1},  {64, 128, 1}, {128, 256, 1}, {256, 512, 1},
      {512, 512, 1},{512, 512, 0},{512, 512, 1},{1024, 512, 1},{512, 256, 1},
      {768, 256, 1},{256, 128, 1},{384, 128, 1},{128, 64, 1},  {192, 64, 1},
      {64, 32, 1},  {32, 8, 0}};

  // ---- map inputs (dict insertion order: x; per layer w,b[,g,bt]; laps r,c,v) ----
  int idx = 0;
  const float* x_in = (const float*)d_in[idx++];
  const float *Wf[17], *Bf[17], *Gf[17], *Tf[17];
  for (int l = 0; l < 17; ++l) {
    Wf[l] = (const float*)d_in[idx++];
    Bf[l] = (const float*)d_in[idx++];
    if (spec[l][2]) { Gf[l] = (const float*)d_in[idx++]; Tf[l] = (const float*)d_in[idx++]; }
    else            { Gf[l] = nullptr; Tf[l] = nullptr; }
  }
  const int* Lcols[6]; const float* Lvals[6];
  for (int i = 0; i < 6; ++i) {
    idx++;                                  // rows: implicit (7 entries per row)
    Lcols[i] = (const int*)d_in[idx++];
    Lvals[i] = (const float*)d_in[idx++];
  }

  // ---- carve workspace ----
  char* wp = (char*)d_ws;
  auto carve = [&](size_t bytes) -> void* {
    void* r = (void*)wp; wp += (bytes + 255) & ~(size_t)255; return r;
  };
  const size_t MAXE = (size_t)40962 * B * 64;   // largest [V,B,C] tensor (elements)
  float* ACT0 = (float*)carve(MAXE * 4);
  float* ACT1 = (float*)carve(MAXE * 4);
  ChebTemps T;
  T.t1    = (float*)carve(MAXE * 4);
  T.g     = (float*)carve(MAXE * 4);
  T.a0    = (unsigned short*)carve(MAXE * 2);
  T.a1    = (unsigned short*)carve(MAXE * 2);
  T.a2    = (unsigned short*)carve(MAXE * 2);
  T.stats = (float*)carve(2 * 512 * 4);
  float* S5 = (float*)carve((size_t)40962 * B * 64 * 4);
  float* S4 = (float*)carve((size_t)10242 * B * 128 * 4);
  float* S3 = (float*)carve((size_t)2562 * B * 256 * 4);
  float* S2 = (float*)carve((size_t)642 * B * 512 * 4);
  float* S1 = (float*)carve((size_t)162 * B * 512 * 4);
  unsigned short* Wt[17];
  for (int l = 0; l < 17; ++l) {
    int Kc = 3 * spec[l][0], N = spec[l][1];
    Wt[l] = (unsigned short*)carve((size_t)Kc * N * 2);
    dim3 wg(ceil_div(Kc, 256), N);
    k_wt_bf16<<<wg, 256, 0, stream>>>(Wf[l], Wt[l], Kc, N);
  }

  // ---- forward ----
  {
    int n = VL[5] * B * 16;
    k_transpose_in<<<ceil_div(n, 256), 256, 0, stream>>>(x_in, ACT0, T.a0, VL[5], B, 16);
  }
  // encoder (icopool = prefix rows; bf16 x0 piggybacks in T.a0)
  run_cheb(stream, ACT0, T.a0, VL[5], B, 16, 32,  Lcols[5], Lvals[5], Wt[0], Bf[0], Gf[0], Tf[0], ACT1, T);
  run_cheb(stream, ACT1, T.a0, VL[5], B, 32, 64,  Lcols[5], Lvals[5], Wt[1], Bf[1], Gf[1], Tf[1], S5,   T);
  run_cheb(stream, S5,   T.a0, VL[4], B, 64, 128, Lcols[4], Lvals[4], Wt[2], Bf[2], Gf[2], Tf[2], S4,   T);
  run_cheb(stream, S4,   T.a0, VL[3], B, 128, 256,Lcols[3], Lvals[3], Wt[3], Bf[3], Gf[3], Tf[3], S3,   T);
  run_cheb(stream, S3,   T.a0, VL[2], B, 256, 512,Lcols[2], Lvals[2], Wt[4], Bf[4], Gf[4], Tf[4], S2,   T);
  run_cheb(stream, S2,   T.a0, VL[1], B, 512, 512,Lcols[1], Lvals[1], Wt[5], Bf[5], Gf[5], Tf[5], S1,   T);
  run_cheb(stream, S1,   T.a0, VL[0], B, 512, 512,Lcols[0], Lvals[0], Wt[6], Bf[6], nullptr, nullptr, ACT0, T);

  // decoder: unpool -> cheb_bn(a) -> concat skip -> cheb_bn(b)
  struct Step { int lvl, Vs, Vd, la, lb; float* skip; int Cskip; };
  const Step steps[4] = {
      {1, VL[0], VL[1], 7,  8,  S1, 512},
      {2, VL[1], VL[2], 9,  10, S2, 512},
      {3, VL[2], VL[3], 11, 12, S3, 256},
      {4, VL[3], VL[4], 13, 14, S4, 128}};
  float* cur = ACT0;      // [42, B, 512] after e0
  for (int si = 0; si < 4; ++si) {
    const Step& st = steps[si];
    int Cin_a = spec[st.la][0];
    int F = B * Cin_a;
    k_unpool<<<ceil_div(st.Vd * F, 256), 256, 0, stream>>>(cur, ACT1, T.a0, st.Vs, st.Vd, F);
    run_cheb(stream, ACT1, T.a0, st.Vd, B, Cin_a, spec[st.la][1],
             Lcols[st.lvl], Lvals[st.lvl], Wt[st.la], Bf[st.la], Gf[st.la], Tf[st.la], ACT0, T);
    int C1 = spec[st.la][1], C2 = st.Cskip;
    k_concat<<<ceil_div(st.Vd * B * (C1 + C2), 256), 256, 0, stream>>>(
        ACT0, st.skip, ACT1, T.a0, st.Vd * B, C1, C2);
    run_cheb(stream, ACT1, T.a0, st.Vd, B, spec[st.lb][0], spec[st.lb][1],
             Lcols[st.lvl], Lvals[st.lvl], Wt[st.lb], Bf[st.lb], Gf[st.lb], Tf[st.lb], ACT0, T);
    cur = ACT0;
  }
  // d5a: unpool 10242 -> 40962, cheb 64->32 (BN) ; writes ACT0 + T.a0
  {
    int F = B * 64;
    k_unpool<<<ceil_div(VL[5] * F, 256), 256, 0, stream>>>(cur, ACT1, T.a0, VL[4], VL[5], F);
    run_cheb(stream, ACT1, T.a0, VL[5], B, 64, 32, Lcols[5], Lvals[5], Wt[15], Bf[15],
             Gf[15], Tf[15], ACT0, T);
  }
  // d5b: cheb 32->8 (no BN) then [V,B,8] -> [B,V,8] into d_out
  run_cheb(stream, ACT0, T.a0, VL[5], B, 32, 8, Lcols[5], Lvals[5], Wt[16], Bf[16],
           nullptr, nullptr, T.g, T);
  {
    int n = VL[5] * B * 8;
    k_final_transpose<<<ceil_div(n, 256), 256, 0, stream>>>(T.g, (float*)d_out, VL[5], B, 8);
  }
}
